// Block_55508157334158
// MI455X (gfx1250) — matrix operand order repaired and checked by reading, not compiled, not run on hardware
//
#include <hip/hip_runtime.h>
#include <math.h>

typedef __bf16 bf16;
typedef __attribute__((ext_vector_type(8)))  bf16  v8bf;
typedef __attribute__((ext_vector_type(16))) bf16  v16bf;
typedef __attribute__((ext_vector_type(8)))  float v8f;

#define DEVI static __device__ __forceinline__

DEVI bf16 f2bf(float f) {
  unsigned u = __float_as_uint(f);
  u += 0x7fffu + ((u >> 16) & 1u);            // round-to-nearest-even
  unsigned short h = (unsigned short)(u >> 16);
  union { unsigned short s; bf16 b; } cv; cv.s = h;
  return cv.b;
}

// ---------------- block reductions (wave32) ----------------
template<int BS>
DEVI float block_reduce_sum(float v) {
  __shared__ float sbuf[BS / 32];
  __shared__ float stot;
  for (int o = 16; o > 0; o >>= 1) v += __shfl_down(v, o, 32);
  int lane = threadIdx.x & 31, w = threadIdx.x >> 5;
  if (lane == 0) sbuf[w] = v;
  __syncthreads();
  if (w == 0) {
    float t = (threadIdx.x < BS / 32) ? sbuf[threadIdx.x] : 0.f;
    for (int o = 16; o > 0; o >>= 1) t += __shfl_down(t, o, 32);
    if (lane == 0) stot = t;
  }
  __syncthreads();
  float r = stot;
  __syncthreads();
  return r;
}

template<int BS>
DEVI float block_reduce_max(float v) {
  __shared__ float sbuf[BS / 32];
  __shared__ float stot;
  for (int o = 16; o > 0; o >>= 1) v = fmaxf(v, __shfl_down(v, o, 32));
  int lane = threadIdx.x & 31, w = threadIdx.x >> 5;
  if (lane == 0) sbuf[w] = v;
  __syncthreads();
  if (w == 0) {
    float t = (threadIdx.x < BS / 32) ? sbuf[threadIdx.x] : -3.0e38f;
    for (int o = 16; o > 0; o >>= 1) t = fmaxf(t, __shfl_down(t, o, 32));
    if (lane == 0) stot = t;
  }
  __syncthreads();
  float r = stot;
  __syncthreads();
  return r;
}

// ---------------- WMMA bf16 NT GEMM:  C[M,N] = A[M,K] * W[N,K]^T ----------------
// 256 threads = 8 waves (WM x WN arrangement). Register-staged global loads with
// double-buffered LDS: next tile's global_load issues before the WMMA block, the
// ds_store (and its s_wait_loadcnt) lands after it, so HBM/L2 latency is hidden
// behind matrix math. Row clamps & tile pointers hoisted out of the K-loop.
template<int BM, int BN, int WM, int WN>
__global__ __launch_bounds__(256)
void gemm_bf16_nt(const bf16* __restrict__ A, int lda, long sA,
                  const bf16* __restrict__ W, int ldw, long sW,
                  float* __restrict__ C, int ldc, long sC,
                  int M, int N, int K)
{
  constexpr int TM = BM / WM, TN = BN / WN;
  constexpr int AM = TM / 16, AN = TN / 16;
  constexpr int LK = 40;                       // padded LDS row (32 + 8 bf16)
  constexpr int NCA = (BM * 4) / 256;          // 16B chunks per thread (A tile)
  constexpr int NCW = (BN * 4) / 256;          // 16B chunks per thread (W tile)

  A += (long)blockIdx.z * sA;
  W += (long)blockIdx.z * sW;
  C += (long)blockIdx.z * sC;

  const int bm = blockIdx.y * BM, bn = blockIdx.x * BN;
  const int tid = threadIdx.x, lane = tid & 31;
  const int wid = tid >> 5;
  const int wm = (wid % WM) * TM, wn = (wid / WM) * TN;
  const int half = lane >> 4, l16 = lane & 15;

  __shared__ alignas(16) bf16 As[2][BM][LK];
  __shared__ alignas(16) bf16 Ws[2][BN][LK];

  // ---- hoisted per-thread tile pointers (row-clamped once) ----
  const bf16* aga[NCA]; int ar[NCA], ac[NCA];
#pragma unroll
  for (int i = 0; i < NCA; i++) {
    const int c = tid + i * 256;
    ar[i] = c >> 2; ac[i] = (c & 3) * 8;
    int rg = bm + ar[i]; if (rg >= M) rg = M - 1;
    aga[i] = A + (long)rg * lda + ac[i];
  }
  const bf16* wga[NCW]; int wr[NCW], wcc[NCW];
#pragma unroll
  for (int i = 0; i < NCW; i++) {
    const int c = tid + i * 256;
    wr[i] = c >> 2; wcc[i] = (c & 3) * 8;
    int rg = bn + wr[i]; if (rg >= N) rg = N - 1;
    wga[i] = W + (long)rg * ldw + wcc[i];
  }

  v8f acc[AM][AN];
#pragma unroll
  for (int i = 0; i < AM; i++)
#pragma unroll
    for (int j = 0; j < AN; j++)
#pragma unroll
      for (int r = 0; r < 8; r++) acc[i][j][r] = 0.f;

  v8bf areg[NCA], wreg[NCW];

  // ---- prologue: tile 0 -> regs -> LDS[0] ----
#pragma unroll
  for (int i = 0; i < NCA; i++) areg[i] = *(const v8bf*)aga[i];
#pragma unroll
  for (int i = 0; i < NCW; i++) wreg[i] = *(const v8bf*)wga[i];
#pragma unroll
  for (int i = 0; i < NCA; i++) *(v8bf*)&As[0][ar[i]][ac[i]] = areg[i];
#pragma unroll
  for (int i = 0; i < NCW; i++) *(v8bf*)&Ws[0][wr[i]][wcc[i]] = wreg[i];
  __syncthreads();

  const int KT = K / 32;
  for (int kt = 0; kt < KT; kt++) {
    const int cur = kt & 1, nxt = cur ^ 1;
    const bool more = (kt + 1 < KT);

    // issue next tile's global loads early (loadcnt pending across the WMMAs)
    if (more) {
      if (kt + 2 < KT) {
#pragma unroll
        for (int i = 0; i < NCA; i++) __builtin_prefetch(aga[i] + 64, 0, 1); // global_prefetch_b8
#pragma unroll
        for (int i = 0; i < NCW; i++) __builtin_prefetch(wga[i] + 64, 0, 1);
      }
#pragma unroll
      for (int i = 0; i < NCA; i++) { areg[i] = *(const v8bf*)(aga[i] + 32); aga[i] += 32; }
#pragma unroll
      for (int i = 0; i < NCW; i++) { wreg[i] = *(const v8bf*)(wga[i] + 32); wga[i] += 32; }
    }

    // ---- compute from LDS[cur]: fragments per CDNA5 16-bit WMMA layouts ----
    v16bf afr[AM], wfr[AN];
#pragma unroll
    for (int i = 0; i < AM; i++) {
      const int r = wm + i * 16 + l16;
      const int ka = half * 8;                 // lanes 0-15: K 0-7/16-23, lanes 16-31: K 8-15/24-31
      union { v16bf v; v8bf h[2]; } u;
      u.h[0] = *(const v8bf*)&As[cur][r][ka];
      u.h[1] = *(const v8bf*)&As[cur][r][ka + 16];
      afr[i] = u.v;
    }
#pragma unroll
    for (int j = 0; j < AN; j++) {
      const int r = wn + j * 16 + l16;
      const int kb = half * 8;                 // lanes 0-15: K 0-7/16-23, lanes 16-31: K 8-15/24-31
      union { v16bf v; v8bf h[2]; } u;
      u.h[0] = *(const v8bf*)&Ws[cur][r][kb];
      u.h[1] = *(const v8bf*)&Ws[cur][r][kb + 16];
      wfr[j] = u.v;
    }
#pragma unroll
    for (int i = 0; i < AM; i++)
#pragma unroll
      for (int j = 0; j < AN; j++)
        acc[i][j] = __builtin_amdgcn_wmma_f32_16x16x32_bf16(
            false, afr[i], false, wfr[j], (short)0, acc[i][j], false, false);

    // ---- drain staged registers into the other LDS buffer ----
    if (more) {
#pragma unroll
      for (int i = 0; i < NCA; i++) *(v8bf*)&As[nxt][ar[i]][ac[i]] = areg[i];
#pragma unroll
      for (int i = 0; i < NCW; i++) *(v8bf*)&Ws[nxt][wr[i]][wcc[i]] = wreg[i];
    }
    __syncthreads();
  }

  // C layout: lane n = l16 is column; VGPR r holds M = r (lanes 0-15) / r+8 (lanes 16-31)
#pragma unroll
  for (int i = 0; i < AM; i++)
#pragma unroll
    for (int j = 0; j < AN; j++) {
      const int col = bn + wn + j * 16 + l16;
      if (col >= N) continue;
#pragma unroll
      for (int r = 0; r < 8; r++) {
        const int row = bm + wm + i * 16 + r + half * 8;
        if (row < M) C[(long)row * ldc + col] = acc[i][j][r];
      }
    }
}

// ---------------- row RMS-norm; optional bf16 output, optional transposed output ----------------
template<bool OBF16, bool TRANS>
__global__ __launch_bounds__(256)
void rms_rows(const float* __restrict__ in, int ldin, void* __restrict__ out,
              int ldout, int rowlen)
{
  const int row = blockIdx.x;
  const float* ip = in + (long)row * ldin;
  float ss = 0.f;
  for (int c = threadIdx.x; c < rowlen; c += 256) { float v = ip[c]; ss += v * v; }
  ss = block_reduce_sum<256>(ss);
  const float inv = rsqrtf(ss / (float)rowlen + 1e-6f);
  for (int c = threadIdx.x; c < rowlen; c += 256) {
    float v = ip[c] * inv;
    long oi = TRANS ? ((long)c * ldout + row) : ((long)row * ldout + c);
    if (OBF16) ((bf16*)out)[oi] = f2bf(v); else ((float*)out)[oi] = v;
  }
}

// ---------------- tea: rope + rms on q,k; extract v transposed; all bf16 ----------------
__global__ __launch_bounds__(64)
void rope_rms_extract(const float* __restrict__ qkv, const float* __restrict__ cosb,
                      const float* __restrict__ sinb,
                      bf16* __restrict__ qo, bf16* __restrict__ ko, bf16* __restrict__ vo,
                      int Bc, int Tc, int Hc)
{
  const int blk = blockIdx.x;                  // (b*T + t)*H + h
  const int h = blk % Hc; const int bt = blk / Hc;
  const int t = bt % Tc;  const int b = bt / Tc;
  const int j = threadIdx.x;                   // 0..63
  const float* row = qkv + (long)bt * 3072 + h * 192;

  const int jj = j & 31;
  const float cs = cosb[t * 32 + jj], sn = sinb[t * 32 + jj];
  const float sgn = (j < 32) ? sn : -sn;
  const int jp = (j < 32) ? j + 32 : j - 32;

  const float qv = row[j],       qp = row[jp];
  const float kv = row[64 + j],  kp = row[64 + jp];
  const float vv = row[128 + j];
  const float qr = qv * cs + qp * sgn;
  const float kr = kv * cs + kp * sgn;

  float qs = block_reduce_sum<64>(qr * qr);
  const float qinv = rsqrtf(qs / 64.f + 1e-6f);
  float ks = block_reduce_sum<64>(kr * kr);
  const float kinv = rsqrtf(ks / 64.f + 1e-6f);

  const long bh = (long)b * Hc + h;
  qo[(bh * Tc + t) * 64 + j] = f2bf(qr * qinv);
  ko[(bh * Tc + t) * 64 + j] = f2bf(kr * kinv);
  vo[(bh * 64 + j) * Tc + t] = f2bf(vv);       // transposed: [b,h,64,T]
}

// ---------------- softmax over rows (L == 1024), optional causal mask + bf16 out ----------------
template<bool CAUSAL, bool OBF16>
__global__ __launch_bounds__(256)
void softmax_rows(const float* __restrict__ in, void* __restrict__ out,
                  int L, int Tdim, float scale)
{
  const int row = blockIdx.x;
  const int t = CAUSAL ? (row % Tdim) : 0;
  const float* ip = in + (long)row * L;
  float vals[4];
  float mx = -3.0e38f;
  int q = 0;
  for (int c = threadIdx.x; c < L; c += 256, q++) {
    float v = ip[c] * scale;
    if (CAUSAL && c > t) v = -1e30f;
    vals[q] = v;
    mx = fmaxf(mx, v);
  }
  mx = block_reduce_max<256>(mx);
  float s = 0.f;
  q = 0;
  for (int c = threadIdx.x; c < L; c += 256, q++) { vals[q] = __expf(vals[q] - mx); s += vals[q]; }
  s = block_reduce_sum<256>(s);
  const float invs = 1.f / s;
  q = 0;
  for (int c = threadIdx.x; c < L; c += 256, q++) {
    float r = vals[q] * invs;
    if (OBF16) ((bf16*)out)[(long)row * L + c] = f2bf(r);
    else       ((float*)out)[(long)row * L + c] = r;
  }
}

// ---------------- AFT sequential scan over T per (b, channel) ----------------
__global__ __launch_bounds__(256)
void aft_scan(const float* __restrict__ qn, const float* __restrict__ kn,
              const float* __restrict__ v, int ldv,
              float* __restrict__ y, int Tdim, int Cdim, int Btot)
{
  const int idx = blockIdx.x * 256 + threadIdx.x;
  if (idx >= Btot * Cdim) return;
  const int c = idx % Cdim, b = idx / Cdim;
  float num = 0.f, den = 0.f;
  for (int t = 0; t < Tdim; t++) {
    const long r = (long)b * Tdim + t;
    const float w = __expf(kn[r * Cdim + c]);
    num += w * v[r * ldv + c];
    den += w;
    const float qv = qn[r * Cdim + c];
    const float sig = 1.f / (1.f + __expf(-qv));
    y[r * Cdim + c] = sig * (num / (den + 1e-6f));
  }
}

// ---------------- elementwise kernels ----------------
__global__ __launch_bounds__(256)
void conv_bf16_k(const float* __restrict__ in, bf16* __restrict__ out, long n) {
  for (long i = (long)blockIdx.x * 256 + threadIdx.x; i < n; i += (long)gridDim.x * 256)
    out[i] = f2bf(in[i]);
}

__global__ __launch_bounds__(256)
void transpose_conv_k(const float* __restrict__ in, bf16* __restrict__ out, int R, int Cc) {
  long i = (long)blockIdx.x * 256 + threadIdx.x;
  if (i < (long)R * Cc) {
    int r = (int)(i / Cc), c = (int)(i % Cc);
    out[(long)c * R + r] = f2bf(in[i]);
  }
}

__global__ __launch_bounds__(256)
void swiglu_k(const float* __restrict__ in, bf16* __restrict__ out, int Fd, long n) {
  for (long i = (long)blockIdx.x * 256 + threadIdx.x; i < n; i += (long)gridDim.x * 256) {
    const long r = i / Fd; const int c = (int)(i % Fd);
    const float u = in[r * 2 * Fd + c];
    const float g = in[r * 2 * Fd + Fd + c];
    out[i] = f2bf(u * g / (1.f + __expf(-g)));
  }
}

__global__ __launch_bounds__(256)
void combine_eltwise_k(const float* __restrict__ t, const float* __restrict__ c,
                       float* __restrict__ z, long n) {
  for (long i = (long)blockIdx.x * 256 + threadIdx.x; i < n; i += (long)gridDim.x * 256) {
    const float cv = c[i];
    const float s = cv / (1.f + __expf(-cv));   // silu(c)
    z[i] = t[i] * s + cv;
  }
}

__global__ __launch_bounds__(256)
void add_inplace_k(float* __restrict__ dst, const float* __restrict__ src, long n) {
  for (long i = (long)blockIdx.x * 256 + threadIdx.x; i < n; i += (long)gridDim.x * 256)
    dst[i] += src[i];
}

__global__ __launch_bounds__(256)
void add_out_k(const float* __restrict__ a, const float* __restrict__ b,
               float* __restrict__ o, long n) {
  for (long i = (long)blockIdx.x * 256 + threadIdx.x; i < n; i += (long)gridDim.x * 256)
    o[i] = a[i] + b[i];
}

// ---------------- global std (w_norm) ----------------
__global__ __launch_bounds__(256)
void stats_k(const float* __restrict__ in, long n, float* __restrict__ cells) {
  float s = 0.f, ss = 0.f;
  for (long i = (long)blockIdx.x * 256 + threadIdx.x; i < n; i += (long)gridDim.x * 256) {
    const float v = in[i]; s += v; ss += v * v;
  }
  s  = block_reduce_sum<256>(s);
  ss = block_reduce_sum<256>(ss);
  if (threadIdx.x == 0) { atomicAdd(&cells[0], s); atomicAdd(&cells[1], ss); }
}

__global__ __launch_bounds__(256)
void wnorm_scale_k(float* __restrict__ w, long n, const float* __restrict__ cells,
                   float coef, bf16* __restrict__ bfout) {
  const double sum = (double)cells[0], ssum = (double)cells[1];
  double var = (ssum - sum * sum / (double)n) / (double)(n - 1);
  if (var < 0.0) var = 0.0;
  const float sc = coef / ((float)sqrt(var) + 1e-8f);
  for (long i = (long)blockIdx.x * 256 + threadIdx.x; i < n; i += (long)gridDim.x * 256) {
    const float v = w[i] * sc;
    w[i] = v;
    if (bfout) bfout[i] = f2bf(v);
  }
}

// ==================================================================================
extern "C" void kernel_launch(void* const* d_in, const int* in_sizes, int n_in,
                              void* d_out, int out_size, void* d_ws, size_t ws_size,
                              hipStream_t stream)
{
  (void)in_sizes; (void)n_in; (void)out_size; (void)ws_size;
  constexpr int Bc = 4, Tc = 1024, Cc = 1024, Hc = 16, Dh = 64;
  constexpr int Mm = Bc * Tc;                   // 4096

  const float* x_in = (const float*)d_in[0];
  const float* cosb = (const float*)d_in[1];
  const float* sinb = (const float*)d_in[2];
  const float* w_oa = (const float*)d_in[3];
  const float* w_t  = (const float*)d_in[4];
  const float* w_f1 = (const float*)d_in[5];
  const float* w_f2 = (const float*)d_in[6];

  char* ws = (char*)d_ws;
  const size_t MBy = 1ull << 20;
  auto Fp = [&](size_t o) { return (float*)(ws + o); };
  auto Bp = [&](size_t o) { return (bf16*)(ws + o); };

  // workspace layout (bytes)
  const size_t G0  = 0;          // 128MB general f32 arena (qkv / scores / sw / ffn)
  const size_t A1  = 128 * MBy;  // 32MB  attn_bf / ug_bf
  const size_t A2  = 160 * MBy;  // 16MB  x (current)
  const size_t A3  = 176 * MBy;  // 16MB  y / ffn tmp
  const size_t A4  = 192 * MBy;  // 8MB   xn_bf / y_bf
  const size_t A5  = 200 * MBy;  // 12MB  wc bf16 (qkv|sw|o packed)
  const size_t A6  = 212 * MBy;  // 24MB  tea q_bf / k_bf / vT_bf
  const size_t R0o = 236 * MBy;  // 24MB  weight ring 0
  const size_t R1o = 260 * MBy;  // 24MB  weight ring 1
  const size_t R2o = 284 * MBy;  // 24MB  weight ring 2
  const size_t WOA = 308 * MBy;  // 4MB   w_oa bf16
  const size_t WF1 = 312 * MBy;  // 16MB  w_ffn_swiglu bf16
  const size_t WF2 = 328 * MBy;  // 8MB   w_ffn_out bf16
  const size_t ST  = 336 * MBy;  // stats cells

  float* xb     = Fp(A2);
  float* yb     = Fp(A3);
  bf16*  xnb    = Bp(A4);
  bf16*  wcb    = Bp(A5);
  float* statsf = Fp(ST);

  auto gemmL = [&](const bf16* A, int lda, long sA, const bf16* Wp, int ldw, long sW,
                   float* Cp, int ldc, long sC, int Mx, int Nx, int Kx, int bz) {
    dim3 g((Nx + 127) / 128, (Mx + 127) / 128, bz);
    gemm_bf16_nt<128, 128, 4, 2><<<g, 256, 0, stream>>>(A, lda, sA, Wp, ldw, sW,
                                                        Cp, ldc, sC, Mx, Nx, Kx);
  };
  auto gemmS = [&](const bf16* A, int lda, long sA, const bf16* Wp, int ldw, long sW,
                   float* Cp, int ldc, long sC, int Mx, int Nx, int Kx, int bz) {
    dim3 g((Nx + 63) / 64, (Mx + 127) / 128, bz);
    gemm_bf16_nt<128, 64, 4, 2><<<g, 256, 0, stream>>>(A, lda, sA, Wp, ldw, sW,
                                                       Cp, ldc, sC, Mx, Nx, Kx);
  };

  // ---- static conversions + copy x ----
  conv_bf16_k<<<1024, 256, 0, stream>>>(w_oa, Bp(WOA), 2048L * 1024);
  conv_bf16_k<<<2048, 256, 0, stream>>>(w_t, wcb, 6144L * 1024);   // wc bf16 (layer 0) == w_t bf16
  conv_bf16_k<<<2048, 256, 0, stream>>>(w_f1, Bp(WF1), 8192L * 1024);
  conv_bf16_k<<<2048, 256, 0, stream>>>(w_f2, Bp(WF2), 1024L * 4096);
  hipMemcpyAsync(xb, x_in, (size_t)Mm * Cc * 4, hipMemcpyDeviceToDevice, stream);

  // ---- produce(x, w_oa, w_t) ----
  rms_rows<true, false><<<Mm, 256, 0, stream>>>(xb, 1024, xnb, 1024, 1024);
  gemmL(xnb, 1024, 0, Bp(WOA), 1024, 0, Fp(G0), 2048, 0, Mm, 2048, 1024, 1);          // oa [4096,2048]
  rms_rows<true, true><<<Mm, 256, 0, stream>>>(Fp(G0), 2048, Bp(G0 + 32 * MBy), 4096, 1024);        // oT bf16 [1024,4096]
  rms_rows<true, true><<<Mm, 256, 0, stream>>>(Fp(G0) + 1024, 2048, Bp(G0 + 40 * MBy), 4096, 1024); // aT
  gemmL(Bp(G0 + 32 * MBy), 4096, 0, Bp(G0 + 40 * MBy), 4096, 0,
        Fp(G0 + 48 * MBy), 1024, 0, 1024, 1024, 4096, 1);                             // y0 = oT a
  softmax_rows<false, false><<<1024, 256, 0, stream>>>(Fp(G0 + 48 * MBy), Fp(G0 + 48 * MBy),
                                                       1024, Tc, 0.03125f);           // /sqrt(1024)
  transpose_conv_k<<<4096, 256, 0, stream>>>(Fp(G0 + 48 * MBy), Bp(G0 + 52 * MBy), 1024, 1024);
  gemmL(wcb, 1024, 0, Bp(G0 + 52 * MBy), 1024, 0, Fp(R0o), 1024, 0, 6144, 1024, 1024, 1); // y = w_t @ y0
  hipMemsetAsync(statsf, 0, 8, stream);
  stats_k<<<1024, 256, 0, stream>>>(Fp(R0o), 6144L * 1024, statsf);
  wnorm_scale_k<<<1024, 256, 0, stream>>>(Fp(R0o), 6144L * 1024, statsf,
                                          1.0f / sqrtf(6144.f), (bf16*)nullptr);

  // ---- shared layer tail: y -> y@w_sw^T -> swiglu -> @w_o^T -> x += ----
  auto tail = [&]() {
    conv_bf16_k<<<2048, 256, 0, stream>>>(yb, xnb, (long)Mm * Cc);                    // y bf16
    gemmL(xnb, 1024, 0, wcb + 3072L * 1024, 1024, 0, Fp(G0), 2048, 0, Mm, 2048, 1024, 1);
    swiglu_k<<<2048, 256, 0, stream>>>(Fp(G0), Bp(A1), 1024, (long)Mm * 1024);
    gemmL(Bp(A1), 1024, 0, wcb + 5120L * 1024, 1024, 0, Fp(G0 + 32 * MBy), 1024, 0, Mm, 1024, 1024, 1);
    add_inplace_k<<<2048, 256, 0, stream>>>(xb, Fp(G0 + 32 * MBy), (long)Mm * Cc);
  };

  auto do_aft = [&]() {
    rms_rows<true, false><<<Mm, 256, 0, stream>>>(xb, 1024, xnb, 1024, 1024);
    gemmL(xnb, 1024, 0, wcb, 1024, 0, Fp(G0), 3072, 0, Mm, 3072, 1024, 1);            // qkv
    rms_rows<false, false><<<Mm, 256, 0, stream>>>(Fp(G0), 3072, Fp(G0 + 64 * MBy), 1024, 1024);        // qn
    rms_rows<false, false><<<Mm, 256, 0, stream>>>(Fp(G0) + 1024, 3072, Fp(G0 + 80 * MBy), 1024, 1024); // kn
    aft_scan<<<16, 256, 0, stream>>>(Fp(G0 + 64 * MBy), Fp(G0 + 80 * MBy),
                                     Fp(G0) + 2048, 3072, yb, Tc, Cc, Bc);
    tail();
  };

  auto do_tea = [&]() {
    rms_rows<true, false><<<Mm, 256, 0, stream>>>(xb, 1024, xnb, 1024, 1024);
    gemmL(xnb, 1024, 0, wcb, 1024, 0, Fp(G0), 3072, 0, Mm, 3072, 1024, 1);            // qkv
    rope_rms_extract<<<Bc * Tc * Hc, 64, 0, stream>>>(Fp(G0), cosb, sinb,
                                                      Bp(A6), Bp(A6 + 8 * MBy), Bp(A6 + 16 * MBy),
                                                      Bc, Tc, Hc);
    for (int b = 0; b < Bc; b++) {
      const bf16* qb = Bp(A6) + (long)b * Hc * Tc * Dh;
      const bf16* kb = Bp(A6 + 8 * MBy) + (long)b * Hc * Tc * Dh;
      const bf16* vb = Bp(A6 + 16 * MBy) + (long)b * Hc * Dh * Tc;
      // scores[h,t,s] = q[t,:] . k[s,:]  -- batched over 16 heads
      gemmL(qb, Dh, (long)Tc * Dh, kb, Dh, (long)Tc * Dh,
            Fp(G0), Tc, (long)Tc * Tc, Tc, Tc, Dh, Hc);
      softmax_rows<true, true><<<Hc * Tc, 256, 0, stream>>>(Fp(G0), Bp(A1), Tc, Tc, 0.125f);
      // y[b,t,h*64+j] = attn[h,t,:] . vT[h,j,:]
      gemmS(Bp(A1), Tc, (long)Tc * Tc, vb, Tc, (long)Dh * Tc,
            yb + (long)b * Tc * Cc, Cc, (long)Dh, Tc, Dh, Tc, Hc);
    }
    tail();
  };

  auto do_combine = [&](const float* tw, const float* cw, float* nx) {
    combine_eltwise_k<<<2048, 256, 0, stream>>>(tw, cw, nx, 6144L * 1024);
    const size_t offs[3] = {0, 3072ul * 1024, 5120ul * 1024};
    const long   cnts[3] = {3072L * 1024, 2048L * 1024, 1024L * 1024};
    for (int k = 0; k < 3; k++) {
      hipMemsetAsync(statsf, 0, 8, stream);
      stats_k<<<512, 256, 0, stream>>>(nx + offs[k], cnts[k], statsf);
      wnorm_scale_k<<<512, 256, 0, stream>>>(nx + offs[k], cnts[k], statsf,
                                             0.03125f /* 1/sqrt(1024) */, wcb + offs[k]);
    }
  };

  // ---- layer loop with wt/wc ring rotation ----
  const float* wt_f = Fp(R0o);    // produce output
  const float* wc_f = w_t;        // layer-0 wc aliases read-only input
  float* sched[4] = {Fp(R1o), Fp(R2o), Fp(R0o), Fp(R1o)};
  for (int i = 0; i < 4; i++) {
    if (i == 3) do_tea(); else do_aft();
    do_combine(wt_f, wc_f, sched[i]);    // also refreshes wcb (A5) bf16
    wt_f = wc_f;
    wc_f = sched[i];
  }

  // ---- final tea + FFN ----
  do_tea();
  rms_rows<true, false><<<Mm, 256, 0, stream>>>(xb, 1024, xnb, 1024, 1024);
  gemmL(xnb, 1024, 0, Bp(WF1), 1024, 0, Fp(G0), 8192, 0, Mm, 8192, 1024, 1);
  swiglu_k<<<4096, 256, 0, stream>>>(Fp(G0), Bp(A1), 4096, (long)Mm * 4096);
  gemmL(Bp(A1), 4096, 0, Bp(WF2), 4096, 0, yb, 1024, 0, Mm, 1024, 4096, 1);
  add_out_k<<<2048, 256, 0, stream>>>(xb, yb, (float*)d_out, (long)Mm * Cc);
}
